// eVONet_80522046866083
// MI455X (gfx1250) — compile-verified, matrix operand order checked by reading, not run on hardware
//
#include <hip/hip_runtime.h>

// ---------------- types for WMMA operand shuttling ----------------
typedef __attribute__((ext_vector_type(16))) __bf16 v16bf;
typedef __attribute__((ext_vector_type(8)))  float  v8f;

struct alignas(16) U128 { unsigned int x0, x1, x2, x3; };
struct alignas(32) U256 { U128 lo, hi; };

// ---------------- problem constants (from setup_inputs) ----------------
#define N_FR 16
#define C_CH 128
#define H0   120
#define W0   160
#define H1   30
#define W1   40
#define M_PT 1536

__device__ __forceinline__ unsigned short f32_to_bf16(float f) {
    unsigned int u = __float_as_uint(f);
    unsigned int r = (u + 0x7FFFu + ((u >> 16) & 1u)) >> 16;   // RNE
    return (unsigned short)r;
}
__device__ __forceinline__ float bf16_to_f32(unsigned short h) {
    return __uint_as_float(((unsigned int)h) << 16);
}

// ---------------- preprocess: NCHW f32 -> NHWC bf16 (level 0) ----------------
__global__ __launch_bounds__(256)
void nchw_to_nhwc_bf16(const float* __restrict__ fm, unsigned short* __restrict__ out)
{
    __shared__ unsigned short tile[C_CH * 32];           // [c][w]
    const int nh = blockIdx.x;                           // n*H0 + h
    const int n  = nh / H0, h = nh % H0;
    const int t  = threadIdx.x;
    for (int w0 = 0; w0 < W0; w0 += 32) {
        __syncthreads();
        #pragma unroll
        for (int it = 0; it < 16; ++it) {                // 128c x 32w / 256
            int idx = it * 256 + t;
            int c = idx >> 5, w = idx & 31;              // 32 consecutive w per c: coalesced
            float f = fm[(((size_t)n * C_CH + c) * H0 + h) * W0 + w0 + w];
            tile[c * 32 + w] = f32_to_bf16(f);
        }
        __syncthreads();
        #pragma unroll
        for (int it = 0; it < 16; ++it) {
            int idx = it * 256 + t;
            int w = idx >> 7, c = idx & 127;             // 128 consecutive c: coalesced
            out[(((size_t)n * H0 + h) * W0 + w0 + w) * C_CH + c] = tile[c * 32 + w];
        }
    }
}

// ---------------- preprocess: 4x4 avg-pool -> level 1 NHWC bf16 ----------------
__global__ __launch_bounds__(128)
void pool4_kernel(const unsigned short* __restrict__ f0, unsigned short* __restrict__ f1)
{
    int idx = blockIdx.x;                 // n*H1*W1 + h4*W1 + w4
    int n = idx / (H1 * W1); int r = idx - n * (H1 * W1);
    int h4 = r / W1, w4 = r - h4 * W1;
    int c = threadIdx.x;
    float s = 0.f;
    #pragma unroll
    for (int i2 = 0; i2 < 4; ++i2)
        #pragma unroll
        for (int j2 = 0; j2 < 4; ++j2)
            s += bf16_to_f32(f0[(((size_t)n * H0 + h4*4 + i2) * W0 + (w4*4 + j2)) * C_CH + c]);
    f1[(((size_t)n * H1 + h4) * W1 + w4) * C_CH + c] = f32_to_bf16(s * 0.0625f);
}

// ---------------- preprocess: gmap [M,C,3,3] f32 -> [M,9,C] bf16 ----------------
__global__ __launch_bounds__(128)
void gmap_kernel(const float* __restrict__ gm, unsigned short* __restrict__ gt)
{
    int m = blockIdx.x, c = threadIdx.x;
    #pragma unroll
    for (int pq = 0; pq < 9; ++pq)
        gt[((size_t)m * 9 + pq) * C_CH + c] = f32_to_bf16(gm[((size_t)m * C_CH + c) * 9 + pq]);
}

// ---------------- main: one wave32 per edge, WMMA bf16 correlation ----------------
__global__ __launch_bounds__(32)
void corr_kernel(const U128* __restrict__ f0, const U128* __restrict__ f1,
                 const U128* __restrict__ gq, const float* __restrict__ coords,
                 const int* __restrict__ ii, const int* __restrict__ jj,
                 float* __restrict__ out, int E)
{
    __shared__ U128  Wlds[112 * 16];     // 112 union rows x 128 bf16 channels (28.0 KB)
    __shared__ float Rlds[112 * 16];     // R[pos][pq] (7.0 KB)
    __shared__ int   pqx[16], pqy[16];
    __shared__ float pqfx[16], pqfy[16];

    const int e = blockIdx.x;
    if (e >= E) return;
    const int lane = threadIdx.x;
    const int rowm = lane & 15;          // M (A) / N (B) index
    const int sel  = lane >> 4;          // K-half select per 16-bit operand layout

    const int iiE = ii[e];
    const int jjE = jj[e];

    // --- B operand: G[pq][c], lanes = pq columns, zero-pad cols 9..15 (level-invariant) ---
    const U128* gb = gq + (size_t)iiE * 9 * 16;
    __builtin_prefetch(gb, 0, 1);        // global_prefetch_b8
    U256 Breg[4];
    #pragma unroll
    for (int s = 0; s < 4; ++s) {
        if (rowm < 9) {
            Breg[s].lo = gb[rowm * 16 + 4*s + sel];         // K = 32s+8sel .. +7
            Breg[s].hi = gb[rowm * 16 + 4*s + sel + 2];     // K = 32s+16+8sel .. +7
        } else {
            Breg[s].lo = U128{0,0,0,0};
            Breg[s].hi = U128{0,0,0,0};
        }
    }

    for (int l = 0; l < 2; ++l) {
        const int   Hl   = l ? H1 : H0;
        const int   Wl   = l ? W1 : W0;
        const float invL = l ? 0.25f : 1.0f;
        const U128* fb   = l ? (f1 + (size_t)jjE * H1 * W1 * 16)
                             : (f0 + (size_t)jjE * H0 * W0 * 16);

        // per-(p,q) integer base + bilinear fractions
        if (lane < 9) {
            float x = coords[((size_t)e * 9 + lane) * 2 + 0] * invL;
            float y = coords[((size_t)e * 9 + lane) * 2 + 1] * invL;
            float x0 = floorf(x), y0 = floorf(y);
            pqx[lane] = (int)x0;  pqy[lane] = (int)y0;
            pqfx[lane] = x - x0;  pqfy[lane] = y - y0;
        }
        __syncthreads();

        int minx0 = pqx[0], miny0 = pqy[0];
        #pragma unroll
        for (int k = 1; k < 9; ++k) { minx0 = min(minx0, pqx[k]); miny0 = min(miny0, pqy[k]); }
        const int wx0 = minx0 - 3, wy0 = miny0 - 3;

        // --- stage clipped 10x10x128 bf16 union window; zero rows 100..111 ---
        for (int it = 0; it < 56; ++it) {
            int pos = 2 * it + sel;                         // lanes 0-15: even, 16-31: odd
            U128 v{0,0,0,0};
            if (pos < 100) {
                int r  = pos / 10, cc = pos - 10 * r;
                int ys = wy0 + r;  ys = ys < 0 ? 0 : (ys > Hl - 1 ? Hl - 1 : ys);
                int xs = wx0 + cc; xs = xs < 0 ? 0 : (xs > Wl - 1 ? Wl - 1 : xs);
                v = fb[((size_t)ys * Wl + xs) * 16 + rowm]; // coalesced 256B/row
            }
            Wlds[pos * 16 + rowm] = v;
        }
        __syncthreads();

        // --- R[112x16] = Window[112x128] x G[128x16] : 7 tiles x 4 K-steps of WMMA ---
        #pragma unroll
        for (int t = 0; t < 7; ++t) {
            v8f acc = {0.f,0.f,0.f,0.f,0.f,0.f,0.f,0.f};
            #pragma unroll
            for (int s = 0; s < 4; ++s) {
                U256 Areg;
                const int base = (16 * t + rowm) * 16 + 4*s + sel;
                Areg.lo = Wlds[base];
                Areg.hi = Wlds[base + 2];
                acc = __builtin_amdgcn_wmma_f32_16x16x32_bf16(
                        false, __builtin_bit_cast(v16bf, Areg),
                        false, __builtin_bit_cast(v16bf, Breg[s]),
                        (short)0, acc, false, false);
            }
            // C/D layout: VGPR j -> (M=j, N=lane) lanes 0-15; (M=j+8, N=lane-16) lanes 16-31
            #pragma unroll
            for (int j = 0; j < 8; ++j)
                Rlds[(16 * t + j + 8 * sel) * 16 + rowm] = acc[j];
        }
        __syncthreads();

        // --- bilinear 2x2 blend epilogue: 49 shifts x 9 patch positions ---
        for (int idx = lane; idx < 441; idx += 32) {
            int pq = idx / 49;
            int ij = idx - 49 * pq;
            int i = ij / 7, j = ij - 7 * i;
            int ry = pqy[pq] - miny0;                       // 0..2
            int rx = pqx[pq] - minx0;
            int base = (ry + i) * 10 + (rx + j);            // <= 99
            float c00 = Rlds[(base     ) * 16 + pq];
            float c01 = Rlds[(base +  1) * 16 + pq];
            float c10 = Rlds[(base + 10) * 16 + pq];
            float c11 = Rlds[(base + 11) * 16 + pq];
            float fx = pqfx[pq], fy = pqfy[pq];
            float v = (1.f - fy) * ((1.f - fx) * c00 + fx * c01)
                    +        fy  * ((1.f - fx) * c10 + fx * c11);
            out[(size_t)e * 882 + (size_t)(ij * 9 + pq) * 2 + l] = v;
        }
        __syncthreads();   // before LDS reuse by next level
    }
}

// ---------------- host launcher ----------------
extern "C" void kernel_launch(void* const* d_in, const int* in_sizes, int n_in,
                              void* d_out, int out_size, void* d_ws, size_t ws_size,
                              hipStream_t stream)
{
    const float* fmap   = (const float*)d_in[0];   // [1,16,128,120,160]
    const float* gmap   = (const float*)d_in[1];   // [1,1536,128,3,3]
    const float* coords = (const float*)d_in[2];   // [1,E,3,3,2]
    const int*   ii     = (const int*)d_in[3];     // [E]
    const int*   jj     = (const int*)d_in[4];     // [E]
    float*       out    = (float*)d_out;
    const int    E      = in_sizes[3];

    // workspace layout (bf16 stored as u16): fmT0 | fmT1 | gmT
    unsigned short* ws   = (unsigned short*)d_ws;
    const size_t fmT0_n  = (size_t)N_FR * H0 * W0 * C_CH;   // 39,321,600
    const size_t fmT1_n  = (size_t)N_FR * H1 * W1 * C_CH;   //  2,457,600
    unsigned short* fmT0 = ws;
    unsigned short* fmT1 = ws + fmT0_n;
    unsigned short* gmT  = ws + fmT0_n + fmT1_n;

    nchw_to_nhwc_bf16<<<N_FR * H0, 256, 0, stream>>>(fmap, fmT0);
    pool4_kernel<<<N_FR * H1 * W1, 128, 0, stream>>>(fmT0, fmT1);
    gmap_kernel<<<M_PT, 128, 0, stream>>>(gmap, gmT);

    corr_kernel<<<E, 32, 0, stream>>>((const U128*)fmT0, (const U128*)fmT1,
                                      (const U128*)gmT, coords, ii, jj, out, E);
}
